// AttentionLayerWithRPR_22368189678180
// MI455X (gfx1250) — compile-verified, matrix operand order checked by reading, not run on hardware
//
#include <hip/hip_runtime.h>

// ---------------------------------------------------------------------------
// AttentionLayerWithRPR for MI455X (gfx1250, wave32, WMMA).
// B=4, S=1024, H=8, DH=64, D=512, NUM_RPR=11.
//
// Roofline: ~24 GFLOP total, required traffic ~50MB -> ~2us @ 23.3TB/s.
// Never materialize alpha (256MB) or scores (134MB): Shaw RPR factorization
// (qk_rpr table + probability buckets) + flash-style streaming through LDS.
// All GEMMs use v_wmma_f32_16x16x32_bf16. bf16 staged to LDS as packed b32.
// Softmax runs in exp2 domain (log2e folded into the 1/sqrt(DH) scale).
// ---------------------------------------------------------------------------

typedef __attribute__((ext_vector_type(16))) __bf16 v16bf;
typedef __attribute__((ext_vector_type(8)))  float  v8f;

#define WMMA_BF16(a, b, c) \
  __builtin_amdgcn_wmma_f32_16x16x32_bf16(false, (a), false, (b), (short)0, (c), false, false)

#define S_   1024
#define D_   512
#define H_   8
#define DH_  64
#define BH_  32    // B*H
#define NRP  11
#define QT   128   // queries per attention block
#define KT   64    // keys per LDS stage
#define SCL  0.18033688f   // (1/sqrt(64)) * log2(e): softmax in exp2 domain

union bfpack { __bf16 h[2]; unsigned u; };

__device__ __forceinline__ unsigned pack2f(float a, float b) {
  bfpack r; r.h[0] = (__bf16)a; r.h[1] = (__bf16)b; return r.u;  // v_cvt_pk_bf16_f32
}
__device__ __forceinline__ unsigned pack2h(__bf16 a, __bf16 b) {
  bfpack r; r.h[0] = a; r.h[1] = b; return r.u;
}

// 16x32 A fragment / 32x16 B fragment (from an N-major tile), CDNA5 7.12.2:
//   lane l: row = l&15, kb = (l<16)?0:8
//   elems 0..7  = T[row][kb+0..7]      elems 8..15 = T[row][kb+16..16+7]
__device__ __forceinline__ v16bf ld_frag(const __bf16* tile, int ld, int lane) {
  const int row = lane & 15;
  const int kb  = (lane & 16) ? 8 : 0;
  const __bf16* p = tile + row * ld + kb;
  v16bf f;
#pragma unroll
  for (int i = 0; i < 8; ++i) { f[i] = p[i]; f[i + 8] = p[i + 16]; }
  return f;
}

// ---------------------------------------------------------------------------
// Kernel 1: projection GEMM  Y[b,h,s,dh] = (X[4096,512] @ W[512,512] + bias)
// 64x64 block tile, 8 waves each doing one 16x32 output strip, K staged 32.
// ---------------------------------------------------------------------------
__global__ __launch_bounds__(256) void proj_kernel(
    const float* __restrict__ X, const float* __restrict__ W,
    const float* __restrict__ bias, __bf16* __restrict__ Y) {
  __shared__ __bf16 As[64 * 32];   // A tile, row-major
  __shared__ __bf16 Bt[64 * 32];   // W tile, N-major (transposed)

  const int tid  = threadIdx.x;
  const int lane = tid & 31;
  const int w    = tid >> 5;
  const int gm   = blockIdx.x * 64;
  const int gn   = blockIdx.y * 64;
  const int wm   = w >> 1;          // 0..3 -> row strip
  const int wn   = (w & 1) * 32;    // 0/32 -> col strip (two 16-wide tiles)

  v8f acc0 = {}, acc1 = {};

  for (int k0 = 0; k0 < D_; k0 += 32) {
    __syncthreads();
#pragma unroll
    for (int i = 0; i < 4; ++i) {       // stage A: 64x32 f32 -> packed bf16x2
      int e = tid + i * 256;            // pair index; row = e>>4, colpair = e&15
      int row = e >> 4, cp = e & 15;
      float2 x2 = *(const float2*)&X[(size_t)(gm + row) * D_ + k0 + 2 * cp];
      ((unsigned*)As)[e] = pack2f(x2.x, x2.y);
    }
#pragma unroll
    for (int i = 0; i < 4; ++i) {       // stage W transposed: Bt[n][k], packed
      int e = tid + i * 256;
      int nn = e & 63, kp = e >> 6;     // kp = k-pair 0..15
      float w0 = W[(size_t)(k0 + 2 * kp)     * D_ + gn + nn];
      float w1 = W[(size_t)(k0 + 2 * kp + 1) * D_ + gn + nn];
      ((unsigned*)Bt)[nn * 16 + kp] = pack2f(w0, w1);
    }
    if (k0 + 32 < D_) {                 // prefetch next K-stage
      __builtin_prefetch(&X[(size_t)(gm + (tid & 63)) * D_ + k0 + 32], 0, 0);
      __builtin_prefetch(&W[(size_t)(k0 + 32 + (tid >> 3)) * D_ + gn], 0, 0);
    }
    __syncthreads();

    v16bf a  = ld_frag(As + wm * 16 * 32, 32, lane);
    v16bf b0 = ld_frag(Bt + wn * 32, 32, lane);
    v16bf b1 = ld_frag(Bt + (wn + 16) * 32, 32, lane);
    acc0 = WMMA_BF16(a, b0, acc0);
    acc1 = WMMA_BF16(a, b1, acc1);
  }

  const int n = lane & 15, hi = lane >> 4;
#pragma unroll
  for (int j = 0; j < 8; ++j) {
    int m    = j + 8 * hi;
    int grow = gm + wm * 16 + m;            // global row = b*S + s
    int b_   = grow >> 10, s_ = grow & 1023;
    int c0   = gn + wn + n, c1 = c0 + 16;   // global col = h*DH + dh
    float v0 = acc0[j] + bias[c0];
    float v1 = acc1[j] + bias[c1];
    Y[(((size_t)b_ * H_ + (c0 >> 6)) * S_ + s_) * DH_ + (c0 & 63)] = (__bf16)v0;
    Y[(((size_t)b_ * H_ + (c1 >> 6)) * S_ + s_) * DH_ + (c1 & 63)] = (__bf16)v1;
  }
}

// ---------------------------------------------------------------------------
// Kernel 2: qk_rpr[bh,s,r] = dot(qh[bh,s,:], krpr[r,:])  (r < 11, padded 16)
// ---------------------------------------------------------------------------
__global__ __launch_bounds__(256) void qkrpr_kernel(
    const __bf16* __restrict__ qh, const float* __restrict__ krpr,
    float* __restrict__ qkr) {
  __shared__ float kl[NRP * DH_];
  for (int i = threadIdx.x; i < NRP * DH_; i += 256) kl[i] = krpr[i];
  __syncthreads();

  const int row = blockIdx.x * 256 + threadIdx.x;     // 0..32767 = bh*S + s
  const v16bf* qp = (const v16bf*)(qh + (size_t)row * DH_);
  float qv[DH_];
#pragma unroll
  for (int c = 0; c < 4; ++c) {
    v16bf t = qp[c];
#pragma unroll
    for (int i = 0; i < 16; ++i) qv[c * 16 + i] = (float)t[i];
  }
#pragma unroll
  for (int r = 0; r < NRP; ++r) {
    float s = 0.f;
#pragma unroll
    for (int d = 0; d < DH_; ++d) s += qv[d] * kl[r * DH_ + d];
    qkr[(size_t)row * 16 + r] = s;
  }
}

// ---------------------------------------------------------------------------
// Kernel 3: flash attention with RPR.
// Block = 8 waves x 16 query rows = 128 queries; each wave owns its rows'
// online softmax (no cross-wave reductions). Keys streamed in 64-wide
// stages (prefetched one stage ahead), processed 32 at a time.
// ---------------------------------------------------------------------------
__global__ __launch_bounds__(256) void attn_kernel(
    const __bf16* __restrict__ qh, const __bf16* __restrict__ kh,
    const __bf16* __restrict__ vh, const float* __restrict__ qkr,
    const int* __restrict__ rpr, const float* __restrict__ krpr,
    float* __restrict__ out) {
  __shared__ __bf16 qs[QT * DH_];       // 16 KB  Q tile [q][d]
  __shared__ __bf16 ks[KT * DH_];       //  8 KB  K tile [k][d]   (N-major for QK^T)
  __shared__ __bf16 vts[DH_ * KT];      //  8 KB  V tile [d][k]   (N-major for P@V)
  __shared__ float  wacc[QT * 16];      //  8 KB  RPR probability buckets
  __shared__ float  qkl[QT * 16];       //  8 KB  qk_rpr slice for this block
  __shared__ float  krl[NRP * DH_];     //  2.75 KB
  __shared__ __bf16 pbuf[8][16 * 32];   //  8 KB  per-wave P staging

  const int tid  = threadIdx.x;
  const int lane = tid & 31;
  const int w    = tid >> 5;
  const int bh   = blockIdx.y;             // 0..31
  const int q0   = blockIdx.x * QT;
  const int b_   = bh >> 3;
  const int h_   = bh & 7;
  const int n    = lane & 15;
  const int hi   = lane >> 4;

  { // stage Q tile, qk_rpr slice, krpr, zero buckets
    const int4* qg = (const int4*)(qh + ((size_t)bh * S_ + q0) * DH_);
    int4* qsv = (int4*)qs;
    for (int i = tid; i < QT * DH_ / 8; i += 256) qsv[i] = qg[i];
    const float4* qrg = (const float4*)(qkr + ((size_t)bh * S_ + q0) * 16);
    float4* qklv = (float4*)qkl;
    for (int i = tid; i < QT * 16 / 4; i += 256) qklv[i] = qrg[i];
    for (int i = tid; i < NRP * DH_; i += 256) krl[i] = krpr[i];
    for (int i = tid; i < QT * 16; i += 256) wacc[i] = 0.f;
  }
  __syncthreads();

  const int wq  = w * 16;        // this wave's local query row base
  const int qg0 = q0 + wq;       // global
  v16bf qa0 = ld_frag(qs + wq * DH_ + 0,  DH_, lane);
  v16bf qa1 = ld_frag(qs + wq * DH_ + 32, DH_, lane);

  v8f o0 = {}, o1 = {}, o2 = {}, o3 = {};
  float mrow[8], lrow[8];
  int   rbase[8];                // hoisted 32-bit gather bases: qrow*S + n
#pragma unroll
  for (int j = 0; j < 8; ++j) {
    mrow[j] = -1e30f; lrow[j] = 0.f;
    rbase[j] = (qg0 + j + 8 * hi) * S_ + n;
  }

  for (int k0 = 0; k0 < S_; k0 += KT) {
    __syncthreads();
    { // stage K (row-major, vectorized) and V (transposed, packed b32)
      const int4* kg = (const int4*)(kh + ((size_t)bh * S_ + k0) * DH_);
      int4* ksv = (int4*)ks;
      for (int i = tid; i < KT * DH_ / 8; i += 256) ksv[i] = kg[i];
      const __bf16* vg = vh + ((size_t)bh * S_ + k0) * DH_;
      for (int i = tid; i < (KT / 2) * DH_; i += 256) {
        int d = i & 63, kp = i >> 6;               // kp = key-pair 0..31
        ((unsigned*)vts)[d * (KT / 2) + kp] =
            pack2h(vg[(2 * kp) * DH_ + d], vg[(2 * kp + 1) * DH_ + d]);
      }
    }
    if (k0 + KT < S_) {   // prefetch next stage (8 KB each) into cache
      const __bf16* nk = kh + ((size_t)bh * S_ + k0 + KT) * DH_;
      const __bf16* nv = vh + ((size_t)bh * S_ + k0 + KT) * DH_;
      __builtin_prefetch(nk + tid * 16, 0, 0);
      __builtin_prefetch(nv + tid * 16, 0, 0);
    }
    __syncthreads();

    for (int ks2 = 0; ks2 < KT; ks2 += 32) {
      const int kc = k0 + ks2;
      // ---- scores for 32 keys: two 16-wide fragments ----
      v8f s0 = {}, s1 = {};
      {
        v16bf b0 = ld_frag(ks + (ks2 + 0)  * DH_ + 0,  DH_, lane);
        v16bf b1 = ld_frag(ks + (ks2 + 0)  * DH_ + 32, DH_, lane);
        s0 = WMMA_BF16(qa0, b0, s0);
        s0 = WMMA_BF16(qa1, b1, s0);
        v16bf b2 = ld_frag(ks + (ks2 + 16) * DH_ + 0,  DH_, lane);
        v16bf b3 = ld_frag(ks + (ks2 + 16) * DH_ + 32, DH_, lane);
        s1 = WMMA_BF16(qa0, b2, s1);
        s1 = WMMA_BF16(qa1, b3, s1);
      }

      // ---- gathered RPR key bias (table in LDS) + exp2-domain scale ----
      int r0[8], r1[8];
#pragma unroll
      for (int j = 0; j < 8; ++j) {
        r0[j] = rpr[rbase[j] + kc];                   // saddr + 32-bit voffset
        r1[j] = rpr[rbase[j] + kc + 16];
        const float* qr = qkl + (wq + j + 8 * hi) * 16;   // LDS gather
        s0[j] = (s0[j] + qr[r0[j]]) * SCL;
        s1[j] = (s1[j] + qr[r1[j]]) * SCL;
      }

      // ---- online softmax (base 2): per-row max across 16-lane halves ----
      float fac[8];
#pragma unroll
      for (int j = 0; j < 8; ++j) {
        float v = fmaxf(s0[j], s1[j]);
#pragma unroll
        for (int msk = 1; msk < 16; msk <<= 1) v = fmaxf(v, __shfl_xor(v, msk, 32));
        float mn = fmaxf(mrow[j], v);
        fac[j] = __builtin_amdgcn_exp2f(mrow[j] - mn);
        mrow[j] = mn;
      }
#pragma unroll
      for (int j = 0; j < 8; ++j) {
        o0[j] *= fac[j]; o1[j] *= fac[j]; o2[j] *= fac[j]; o3[j] *= fac[j];
        lrow[j] *= fac[j];
        // rescale this wave's buckets: lane handles bucket n, row j+8*hi
        wacc[(wq + j + 8 * hi) * 16 + n] *= fac[j];
      }

      // ---- probabilities: running sum, bucket atomics, P staging ----
#pragma unroll
      for (int j = 0; j < 8; ++j) {
        float p0 = __builtin_amdgcn_exp2f(s0[j] - mrow[j]);
        float p1 = __builtin_amdgcn_exp2f(s1[j] - mrow[j]);
        float v = p0 + p1;
#pragma unroll
        for (int msk = 1; msk < 16; msk <<= 1) v += __shfl_xor(v, msk, 32);
        lrow[j] += v;
        int m = j + 8 * hi;
        atomicAdd(&wacc[(wq + m) * 16 + r0[j]], p0);   // ds_add_f32
        atomicAdd(&wacc[(wq + m) * 16 + r1[j]], p1);
        pbuf[w][m * 32 + n]      = (__bf16)p0;
        pbuf[w][m * 32 + 16 + n] = (__bf16)p1;
      }

      // ---- O += P @ V over the 4 dh-tiles ----
      v16bf pa = ld_frag(pbuf[w], 32, lane);
      v16bf vb;
      vb = ld_frag(vts + 0 * 16 * KT + ks2, KT, lane); o0 = WMMA_BF16(pa, vb, o0);
      vb = ld_frag(vts + 1 * 16 * KT + ks2, KT, lane); o1 = WMMA_BF16(pa, vb, o1);
      vb = ld_frag(vts + 2 * 16 * KT + ks2, KT, lane); o2 = WMMA_BF16(pa, vb, o2);
      vb = ld_frag(vts + 3 * 16 * KT + ks2, KT, lane); o3 = WMMA_BF16(pa, vb, o3);
    }
  }
  __syncthreads();   // fence LDS bucket atomics before the final read

  // ---- finalize: ctx = (O + wacc @ krpr) / l ; write [B,S,D] fp32 ----
#pragma unroll
  for (int j = 0; j < 8; ++j) {
    int m = j + 8 * hi;
    int qrow = qg0 + m;
    float inv = 1.f / lrow[j];
    float rs0 = 0.f, rs1 = 0.f, rs2 = 0.f, rs3 = 0.f;
#pragma unroll
    for (int r = 0; r < NRP; ++r) {
      float wv = wacc[(wq + m) * 16 + r];
      rs0 += wv * krl[r * DH_ + 0 * 16 + n];
      rs1 += wv * krl[r * DH_ + 1 * 16 + n];
      rs2 += wv * krl[r * DH_ + 2 * 16 + n];
      rs3 += wv * krl[r * DH_ + 3 * 16 + n];
    }
    size_t ob = ((size_t)b_ * S_ + qrow) * D_ + h_ * DH_;
    out[ob + 0 * 16 + n] = (o0[j] + rs0) * inv;
    out[ob + 1 * 16 + n] = (o1[j] + rs1) * inv;
    out[ob + 2 * 16 + n] = (o2[j] + rs2) * inv;
    out[ob + 3 * 16 + n] = (o3[j] + rs3) * inv;
  }
}

// ---------------------------------------------------------------------------
extern "C" void kernel_launch(void* const* d_in, const int* in_sizes, int n_in,
                              void* d_out, int out_size, void* d_ws, size_t ws_size,
                              hipStream_t stream) {
  const float* q    = (const float*)d_in[0];
  const float* k    = (const float*)d_in[1];
  const float* v    = (const float*)d_in[2];
  const int*   rpr  = (const int*)  d_in[3];
  const float* wq_k = (const float*)d_in[4];
  const float* wq_b = (const float*)d_in[5];
  const float* wk_k = (const float*)d_in[6];
  const float* wk_b = (const float*)d_in[7];
  const float* wv_k = (const float*)d_in[8];
  const float* wv_b = (const float*)d_in[9];
  const float* krpr = (const float*)d_in[10];
  float* out = (float*)d_out;

  // workspace: qh/kh/vh bf16 [BH,S,DH] (4MB each) + qk_rpr f32 [BH,S,16] (2MB)
  const size_t NE = (size_t)BH_ * S_ * DH_;       // 2,097,152
  __bf16* qh = (__bf16*)d_ws;
  __bf16* kh = qh + NE;
  __bf16* vh = kh + NE;
  float*  qkr = (float*)(vh + NE);

  dim3 gA(64, 8);                                  // M/64 x N/64
  proj_kernel<<<gA, 256, 0, stream>>>(q, wq_k, wq_b, qh);
  proj_kernel<<<gA, 256, 0, stream>>>(k, wk_k, wk_b, kh);
  proj_kernel<<<gA, 256, 0, stream>>>(v, wv_k, wv_b, vh);

  qkrpr_kernel<<<BH_ * S_ / 256, 256, 0, stream>>>(qh, krpr, qkr);

  dim3 gC(S_ / QT, BH_);                           // 8 x 32
  attn_kernel<<<gC, 256, 0, stream>>>(qh, kh, vh, qkr, rpr, krpr, out);
}
